// SALayerCore_75849122448244
// MI455X (gfx1250) — matrix operand order repaired and checked by reading, not compiled, not run on hardware
//
#include <hip/hip_runtime.h>
#include <hip/hip_bf16.h>

// ---------------------------------------------------------------------------
// PointNet++ SA layer for MI455X (gfx1250, wave32, WMMA, async-LDS).
// GEMM: block = 256 thr = 8 waves owns a 128-row M strip, staged to LDS via
// GLOBAL_LOAD_ASYNC_TO_LDS_B128 (ASYNCcnt). Each wave computes 16 rows x N
// cols: one LDS A-fragment feeds N/16 v_wmma_f32_16x16x32_f16.
// BN statistics reduced through LDS atomics, then one global atomic/channel.
// ---------------------------------------------------------------------------

typedef __attribute__((ext_vector_type(16))) _Float16 v16h;
typedef __attribute__((ext_vector_type(8)))  _Float16 v8h;
typedef __attribute__((ext_vector_type(8)))  float    v8f;

#define BATCH 16
#define NPTS  4096
#define CIN   64
#define NPOINT 1024
#define NSAMP 32
#define K0P   96          // 3 + 64 = 67 padded to multiple of 32
#define N0    64
#define N1    64
#define N2    128
#define ROWS  (BATCH * NPOINT * NSAMP)   // 524288
#define RAD2  0.16f
#define BNEPS 0.001f

#if defined(__HIP_DEVICE_COMPILE__) && \
    __has_builtin(__builtin_amdgcn_global_load_async_to_lds_b128)
#define HAVE_ASYNC_LDS 1
typedef int v4i_ __attribute__((vector_size(16)));
typedef __attribute__((address_space(1))) v4i_ As1V4i;   // global b128 chunk
typedef __attribute__((address_space(3))) v4i_ As3V4i;   // LDS b128 chunk
#else
#define HAVE_ASYNC_LDS 0
#endif

// ---------------------------------------------------------------- FPS ------
__global__ __launch_bounds__(256)
void fps_kernel(const float* __restrict__ xyz,
                float* __restrict__ newxyz_ws,
                float* __restrict__ out_newxyz)
{
    __shared__ float dists[NPTS];
    __shared__ float rv[256];
    __shared__ int   ri[256];
    __shared__ float fx, fy, fz;

    const int b   = blockIdx.x;
    const int tid = threadIdx.x;
    const float* px = xyz + (size_t)b * NPTS * 3;

    for (int i = tid; i < NPTS; i += 256) dists[i] = 1e10f;
    __syncthreads();

    int far = 0;
    for (int j = 0; j < NPOINT; ++j) {
        if (tid == 0) {
            float x = px[far * 3 + 0], y = px[far * 3 + 1], z = px[far * 3 + 2];
            fx = x; fy = y; fz = z;
            size_t o = ((size_t)b * NPOINT + j) * 3;
            newxyz_ws[o + 0] = x; newxyz_ws[o + 1] = y; newxyz_ws[o + 2] = z;
            out_newxyz[o + 0] = x; out_newxyz[o + 1] = y; out_newxyz[o + 2] = z;
        }
        __syncthreads();

        float bd = -1.0f; int bi = 0;
        for (int i = tid; i < NPTS; i += 256) {
            float dx = px[i * 3 + 0] - fx;
            float dy = px[i * 3 + 1] - fy;
            float dz = px[i * 3 + 2] - fz;
            float d  = dx * dx + dy * dy + dz * dz;
            float m  = fminf(dists[i], d);
            dists[i] = m;
            if (m > bd) { bd = m; bi = i; }
        }
        rv[tid] = bd; ri[tid] = bi;
        __syncthreads();
        for (int s = 128; s > 0; s >>= 1) {
            if (tid < s) {
                if (rv[tid + s] > rv[tid] ||
                    (rv[tid + s] == rv[tid] && ri[tid + s] < ri[tid])) {
                    rv[tid] = rv[tid + s]; ri[tid] = ri[tid + s];
                }
            }
            __syncthreads();
        }
        far = ri[0];
        __syncthreads();
    }
}

// ---------------------------------------------------------- ball query -----
__global__ __launch_bounds__(64)
void ballq_kernel(const float* __restrict__ xyz,
                  const float* __restrict__ newxyz,
                  int* __restrict__ idx_ws,
                  float* __restrict__ out_idx)
{
    int r = blockIdx.x * blockDim.x + threadIdx.x;   // center id in [0, B*P)
    if (r >= BATCH * NPOINT) return;
    int b = r / NPOINT;
    const float* px = xyz + (size_t)b * NPTS * 3;
    float cx = newxyz[r * 3 + 0], cy = newxyz[r * 3 + 1], cz = newxyz[r * 3 + 2];

    int buf[NSAMP];
    int cnt = 0;
    for (int i = 0; i < NPTS && cnt < NSAMP; ++i) {
        float dx = px[i * 3 + 0] - cx;
        float dy = px[i * 3 + 1] - cy;
        float dz = px[i * 3 + 2] - cz;
        if (dx * dx + dy * dy + dz * dz < RAD2) buf[cnt++] = i;
    }
    int first = (cnt > 0) ? buf[0] : 0;
    for (int t = cnt; t < NSAMP; ++t) buf[t] = first;
    for (int t = 0; t < NSAMP; ++t) {
        idx_ws[(size_t)r * NSAMP + t]  = buf[t];
        out_idx[(size_t)r * NSAMP + t] = (float)buf[t];
    }
}

// ---------------------------------------------------- gather + concat ------
__global__ __launch_bounds__(256)
void gather_kernel(const float* __restrict__ xyz,
                   const float* __restrict__ points,
                   const float* __restrict__ newxyz,
                   const int* __restrict__ idx_ws,
                   _Float16* __restrict__ X)
{
    int r = blockIdx.x * blockDim.x + threadIdx.x;   // row in [0, ROWS)
    if (r >= ROWS) return;
    int bp = r / NSAMP;
    int b  = bp / NPOINT;
    int i  = idx_ws[r];
    const float* pxyz = xyz    + ((size_t)b * NPTS + i) * 3;
    const float* ppts = points + ((size_t)b * NPTS + i) * CIN;
    const float* ctr  = newxyz + (size_t)bp * 3;
    _Float16* row = X + (size_t)r * K0P;
    row[0] = (_Float16)(pxyz[0] - ctr[0]);
    row[1] = (_Float16)(pxyz[1] - ctr[1]);
    row[2] = (_Float16)(pxyz[2] - ctr[2]);
    #pragma unroll 8
    for (int c = 0; c < CIN; ++c) row[3 + c] = (_Float16)ppts[c];
    for (int c = 3 + CIN; c < K0P; ++c) row[c] = (_Float16)0.0f;
}

// ---------------------------------------- weights -> f16, transposed NxK ---
__global__ __launch_bounds__(256)
void pack_weights_kernel(const float* __restrict__ W0,
                         const float* __restrict__ W1,
                         const float* __restrict__ W2,
                         _Float16* __restrict__ Wt0,
                         _Float16* __restrict__ Wt1,
                         _Float16* __restrict__ Wt2)
{
    int t = blockIdx.x * blockDim.x + threadIdx.x;
    if (t < N0 * K0P) {                       // Wt0[n][k], k<67 real else 0
        int n = t / K0P, k = t % K0P;
        Wt0[t] = (_Float16)((k < 67) ? W0[k * N0 + n] : 0.0f);
        return;
    }
    t -= N0 * K0P;
    if (t >= 0 && t < N1 * 64) {
        int n = t / 64, k = t % 64;
        Wt1[t] = (_Float16)W1[k * N1 + n];
        return;
    }
    t -= N1 * 64;
    if (t >= 0 && t < N2 * 64) {
        int n = t / 64, k = t % 64;
        Wt2[t] = (_Float16)W2[k * N2 + n];
    }
}

// ------------------------------------------------------------ utilities ----
__global__ void zero_stats_kernel(float* __restrict__ s, int n)
{
    int t = blockIdx.x * blockDim.x + threadIdx.x;
    if (t < n) s[t] = 0.0f;
}

// ------------------------------------------- WMMA GEMM + fused BN stats ----
// Block (8 waves) owns a 128-row strip of A staged in LDS (async copy).
// Wave w computes rows [strip + 16w, +16) x all N columns: NT = N/16
// accumulator tiles per wave; each A fragment (LDS) feeds NT WMMAs.
// Bt is N x K row-major (weights transposed), tiny -> L0/L2 resident.
template <int N, int K>
__global__ __launch_bounds__(256)
void wmma_gemm_stats(const _Float16* __restrict__ A,
                     const _Float16* __restrict__ Bt,
                     _Float16* __restrict__ Y,
                     float* __restrict__ bn_sum,
                     float* __restrict__ bn_sqs)
{
    constexpr int NT = N / 16;          // accumulator tiles per wave
    constexpr int CH = 128 * K / 8;     // 16B chunks in the A strip
    __shared__ __align__(16) _Float16 sA[128 * K];
    __shared__ float s_sum[N];
    __shared__ float s_sqs[N];

    const int tid = threadIdx.x;
    if (tid < N) { s_sum[tid] = 0.0f; s_sqs[tid] = 0.0f; }

    const _Float16* gsrc = A + (size_t)blockIdx.x * 128 * K;

#if HAVE_ASYNC_LDS
    for (int c = tid; c < CH; c += 256) {
        __builtin_amdgcn_global_load_async_to_lds_b128(
            (As1V4i*)(gsrc + (size_t)c * 8),
            (As3V4i*)(sA + c * 8), 0, 0);
    }
#if __has_builtin(__builtin_amdgcn_s_wait_asynccnt)
    __builtin_amdgcn_s_wait_asynccnt(0);
#else
    asm volatile("s_wait_asynccnt 0x0" ::: "memory");
#endif
#else
    for (int c = tid; c < CH; c += 256) {
        *(v8h*)(sA + c * 8) = *(const v8h*)(gsrc + (size_t)c * 8);
    }
#endif
    __syncthreads();

    const int wave   = tid >> 5;
    const int lane   = tid & 31;
    const int half16 = lane >> 4;       // 0: lanes 0-15, 1: lanes 16-31
    const int idx16  = lane & 15;

    // A fragment source: K chunks {h*8..h*8+7} and {16+h*8..16+h*8+7}
    const _Float16* aRow = sA + (size_t)(wave * 16 + idx16) * K + half16 * 8;

    v8f acc[NT];
    #pragma unroll
    for (int nt = 0; nt < NT; ++nt) acc[nt] = (v8f){};

    #pragma unroll
    for (int kb = 0; kb < K; kb += 32) {
        v8h alo = *(const v8h*)(aRow + kb);          // ds_load_b128
        v8h ahi = *(const v8h*)(aRow + kb + 16);
        v16h a  = __builtin_shufflevector(alo, ahi,
                    0,1,2,3,4,5,6,7,8,9,10,11,12,13,14,15);
        #pragma unroll
        for (int nt = 0; nt < NT; ++nt) {
            // B fragment: column n of W = row n of Bt, K chunks {h*8..h*8+7} and {16+h*8..16+h*8+7}
            const _Float16* bRow = Bt + (size_t)(nt * 16 + idx16) * K + kb + half16 * 8;
            v16h b = __builtin_shufflevector(*(const v8h*)bRow, *(const v8h*)(bRow + 16),
                        0,1,2,3,4,5,6,7,8,9,10,11,12,13,14,15);
            acc[nt] = __builtin_amdgcn_wmma_f32_16x16x32_f16(
                          false, a, false, b, (short)0, acc[nt], false, false);
        }
    }

    // D layout: lane holds col = nt*16 + idx16, rows m = strip + w*16 + h*8 + r
    const size_t mbase = (size_t)blockIdx.x * 128 + wave * 16 + half16 * 8;
    #pragma unroll
    for (int nt = 0; nt < NT; ++nt) {
        const int col = nt * 16 + idx16;
        float s1 = 0.0f, s2 = 0.0f;
        #pragma unroll
        for (int r = 0; r < 8; ++r) {
            float v = acc[nt][r];
            Y[(mbase + r) * N + col] = (_Float16)v;
            s1 += v;
            s2 += v * v;
        }
        atomicAdd(&s_sum[col], s1);                  // ds_add_f32
        atomicAdd(&s_sqs[col], s2);
    }
    __syncthreads();
    if (tid < N) {
        atomicAdd(&bn_sum[tid], s_sum[tid]);         // 1 global atomic/chan
        atomicAdd(&bn_sqs[tid], s_sqs[tid]);
    }
}

// ------------------------------------------------------------ BN pieces ----
__global__ void bn_finalize_kernel(const float* __restrict__ bn_sum,
                                   const float* __restrict__ bn_sqs,
                                   const float* __restrict__ g,
                                   const float* __restrict__ be,
                                   float* __restrict__ scale,
                                   float* __restrict__ shift,
                                   int n, float cnt)
{
    int c = blockIdx.x * blockDim.x + threadIdx.x;
    if (c >= n) return;
    float m  = bn_sum[c] / cnt;
    float v  = bn_sqs[c] / cnt - m * m;
    float sc = g[c] * rsqrtf(v + BNEPS);
    scale[c] = sc;
    shift[c] = be[c] - m * sc;
}

__global__ __launch_bounds__(256)
void bn_apply_kernel(const _Float16* __restrict__ Y,
                     _Float16* __restrict__ X,
                     const float* __restrict__ scale,
                     const float* __restrict__ shift,
                     int n)
{
    size_t t = (size_t)blockIdx.x * blockDim.x + threadIdx.x;
    if (t >= (size_t)ROWS * n) return;
    int c = (int)(t % n);
    float x = (float)Y[t] * scale[c] + shift[c];
    X[t] = (_Float16)fmaxf(x, 0.0f);
}

__global__ __launch_bounds__(256)
void bn_max_kernel(const _Float16* __restrict__ Y,
                   const float* __restrict__ scale,
                   const float* __restrict__ shift,
                   float* __restrict__ out)
{
    int t = blockIdx.x * blockDim.x + threadIdx.x;   // over B*P*N2
    if (t >= BATCH * NPOINT * N2) return;
    int c  = t % N2;
    int bp = t / N2;
    float sc = scale[c], sh = shift[c];
    float m = 0.0f;                                  // relu => max >= 0
    #pragma unroll 4
    for (int s = 0; s < NSAMP; ++s) {
        float y = (float)Y[((size_t)bp * NSAMP + s) * N2 + c];
        m = fmaxf(m, fmaxf(y * sc + sh, 0.0f));
    }
    out[t] = m;
}

// ------------------------------------------------------------------ host ---
static inline size_t align256(size_t x) { return (x + 255) & ~(size_t)255; }

extern "C" void kernel_launch(void* const* d_in, const int* in_sizes, int n_in,
                              void* d_out, int out_size, void* d_ws, size_t ws_size,
                              hipStream_t stream)
{
    (void)in_sizes; (void)n_in; (void)out_size; (void)ws_size;

    const float* xyz    = (const float*)d_in[0];
    const float* points = (const float*)d_in[1];
    const float* W0 = (const float*)d_in[2];
    const float* g0 = (const float*)d_in[4];
    const float* be0 = (const float*)d_in[5];
    const float* W1 = (const float*)d_in[6];
    const float* g1 = (const float*)d_in[8];
    const float* be1 = (const float*)d_in[9];
    const float* W2 = (const float*)d_in[10];
    const float* g2 = (const float*)d_in[12];
    const float* be2 = (const float*)d_in[13];
    // biases b0/b1/b2 cancel exactly inside BN (mean subtraction) -> skipped.

    // Workspace layout
    char* ws = (char*)d_ws;
    size_t off = 0;
    float* ws_newxyz = (float*)(ws + off);
    off = align256(off + (size_t)BATCH * NPOINT * 3 * sizeof(float));
    int* ws_idx = (int*)(ws + off);
    off = align256(off + (size_t)BATCH * NPOINT * NSAMP * sizeof(int));
    float* ws_stats = (float*)(ws + off);          // sum|sqs|scale|shift x128
    off = align256(off + 4 * 128 * sizeof(float));
    _Float16* Wt0 = (_Float16*)(ws + off);
    off = align256(off + (size_t)N0 * K0P * sizeof(_Float16));
    _Float16* Wt1 = (_Float16*)(ws + off);
    off = align256(off + (size_t)N1 * 64 * sizeof(_Float16));
    _Float16* Wt2 = (_Float16*)(ws + off);
    off = align256(off + (size_t)N2 * 64 * sizeof(_Float16));
    _Float16* Xh = (_Float16*)(ws + off);          // activations (reused)
    off = align256(off + (size_t)ROWS * K0P * sizeof(_Float16));
    _Float16* Yh = (_Float16*)(ws + off);          // GEMM outputs (reused)
    off = align256(off + (size_t)ROWS * N2 * sizeof(_Float16));

    float* bn_sum   = ws_stats;
    float* bn_sqs   = ws_stats + 128;
    float* bn_scale = ws_stats + 256;
    float* bn_shift = ws_stats + 384;

    // Output layout: new_xyz | new_points | idx (all float32)
    float* out = (float*)d_out;
    float* out_newxyz = out;
    float* out_newpts = out + (size_t)BATCH * NPOINT * 3;
    float* out_idx    = out + (size_t)BATCH * NPOINT * 3
                            + (size_t)BATCH * NPOINT * N2;

    const float cnt = (float)ROWS;
    const int strips = ROWS / 128;                 // 4096, exact

    // 1. FPS + new_xyz
    fps_kernel<<<BATCH, 256, 0, stream>>>(xyz, ws_newxyz, out_newxyz);
    // 2. Ball query
    ballq_kernel<<<(BATCH * NPOINT + 63) / 64, 64, 0, stream>>>(
        xyz, ws_newxyz, ws_idx, out_idx);
    // 3. Weight prepack (f16, transposed, padded)
    pack_weights_kernel<<<(N0 * K0P + N1 * 64 + N2 * 64 + 255) / 256, 256, 0,
                          stream>>>(W0, W1, W2, Wt0, Wt1, Wt2);
    // 4. Gather + concat -> X0 (ROWS x 96 f16)
    gather_kernel<<<(ROWS + 255) / 256, 256, 0, stream>>>(
        xyz, points, ws_newxyz, ws_idx, Xh);

    // Stage 0: (ROWS x 96) * (96 x 64)
    {
        zero_stats_kernel<<<1, 256, 0, stream>>>(ws_stats, 256);
        wmma_gemm_stats<N0, K0P><<<strips, 256, 0, stream>>>(
            Xh, Wt0, Yh, bn_sum, bn_sqs);
        bn_finalize_kernel<<<1, N0, 0, stream>>>(
            bn_sum, bn_sqs, g0, be0, bn_scale, bn_shift, N0, cnt);
        size_t elems = (size_t)ROWS * N0;
        bn_apply_kernel<<<(unsigned)((elems + 255) / 256), 256, 0, stream>>>(
            Yh, Xh, bn_scale, bn_shift, N0);
    }
    // Stage 1: (ROWS x 64) * (64 x 64)
    {
        zero_stats_kernel<<<1, 256, 0, stream>>>(ws_stats, 256);
        wmma_gemm_stats<N1, 64><<<strips, 256, 0, stream>>>(
            Xh, Wt1, Yh, bn_sum, bn_sqs);
        bn_finalize_kernel<<<1, N1, 0, stream>>>(
            bn_sum, bn_sqs, g1, be1, bn_scale, bn_shift, N1, cnt);
        size_t elems = (size_t)ROWS * N1;
        bn_apply_kernel<<<(unsigned)((elems + 255) / 256), 256, 0, stream>>>(
            Yh, Xh, bn_scale, bn_shift, N1);
    }
    // Stage 2: (ROWS x 64) * (64 x 128) -> BN -> ReLU -> max over samples
    {
        zero_stats_kernel<<<1, 256, 0, stream>>>(ws_stats, 256);
        wmma_gemm_stats<N2, 64><<<strips, 256, 0, stream>>>(
            Xh, Wt2, Yh, bn_sum, bn_sqs);
        bn_finalize_kernel<<<1, N2, 0, stream>>>(
            bn_sum, bn_sqs, g2, be2, bn_scale, bn_shift, N2, cnt);
        bn_max_kernel<<<(BATCH * NPOINT * N2 + 255) / 256, 256, 0, stream>>>(
            Yh, bn_scale, bn_shift, out_newpts);
    }
}
